// Encoder_62569083568687
// MI455X (gfx1250) — matrix operand order repaired and checked by reading, not compiled, not run on hardware
//
#include <hip/hip_runtime.h>
#include <hip/hip_bf16.h>
#include <math.h>

#define Bsz 64
#define Lsz 197
#define Dsz 768
#define Hsz 12
#define NLsz 12
#define Fsz 3072
#define HDsz 64
#define Msz (Bsz * Lsz) /* 12608 */
#define KPAD 224        /* key padding for V^T LDS tile, multiple of 32 */

typedef _Float16 v16h __attribute__((ext_vector_type(16)));
typedef _Float16 v8h  __attribute__((ext_vector_type(8)));
typedef _Float16 v4h  __attribute__((ext_vector_type(4)));
typedef float    v8f  __attribute__((ext_vector_type(8)));

static __device__ __forceinline__ v16h frag_join(v8h lo, v8h hi) {
  return __builtin_shufflevector(lo, hi, 0,1,2,3,4,5,6,7,8,9,10,11,12,13,14,15);
}

static __device__ __forceinline__ v8f wmma_f16(v16h a, v16h b, v8f c) {
  // (neg_a, A, neg_b, B, c_mod, C, reuse_a, reuse_b) -> v_wmma_f32_16x16x32_f16
  return __builtin_amdgcn_wmma_f32_16x16x32_f16(false, a, false, b, (short)0, c,
                                                false, false);
}

// Async global->LDS copy of one 16-byte chunk (CDNA5 VGLOBAL async op,
// tracked by ASYNCcnt).  lds_off = 32-bit LDS offset (flat addr low bits).
static __device__ __forceinline__ void async_copy_b128(unsigned lds_off,
                                                       const void* gsrc) {
  asm volatile("global_load_async_to_lds_b128 %0, %1, off"
               :
               : "v"(lds_off), "v"(gsrc)
               : "memory");
}
static __device__ __forceinline__ void wait_async0() {
  asm volatile("s_wait_asynccnt 0" ::: "memory");
}

// ---------------------------------------------------------------------------
// h = x + pos_emb  (broadcast pos over batch)
// ---------------------------------------------------------------------------
__global__ void addpos_kernel(const float* __restrict__ x,
                              const float* __restrict__ pos,
                              float* __restrict__ h) {
  size_t i = (size_t)blockIdx.x * 256 + threadIdx.x;
  if (i < (size_t)Msz * Dsz) h[i] = x[i] + pos[i % ((size_t)Lsz * Dsz)];
}

// ---------------------------------------------------------------------------
// LayerNorm over D=768, one wave32 per row (24 elements per lane)
// ---------------------------------------------------------------------------
__global__ __launch_bounds__(256) void ln_kernel(const float* __restrict__ X,
                                                 const float* __restrict__ g,
                                                 const float* __restrict__ bb,
                                                 float* __restrict__ Y, int rows) {
  int lane = threadIdx.x & 31, wid = threadIdx.x >> 5;
  int row = blockIdx.x * 8 + wid;
  if (row >= rows) return;
  const float* x = X + (size_t)row * Dsz;
  float v[24];
  float s = 0.f, s2 = 0.f;
#pragma unroll
  for (int j = 0; j < 24; j++) {
    v[j] = x[lane + j * 32];
    s += v[j];
    s2 += v[j] * v[j];
  }
#pragma unroll
  for (int m = 16; m >= 1; m >>= 1) {
    s += __shfl_xor(s, m, 32);
    s2 += __shfl_xor(s2, m, 32);
  }
  float mean = s * (1.f / 768.f);
  float var = s2 * (1.f / 768.f) - mean * mean;
  float rs = rsqrtf(var + 1e-6f);
  float* y = Y + (size_t)row * Dsz;
#pragma unroll
  for (int j = 0; j < 24; j++) {
    int d = lane + j * 32;
    y[d] = (v[j] - mean) * rs * g[d] + bb[d];
  }
}

// ---------------------------------------------------------------------------
// Weight convert+transpose: W[K,N] f32 row-major -> WT[N,K] f16 row-major
// ---------------------------------------------------------------------------
__global__ void wconv_kernel(const float* __restrict__ W,
                             _Float16* __restrict__ WT, int K, int N) {
  size_t i = (size_t)blockIdx.x * 256 + threadIdx.x;
  if (i < (size_t)K * N) {
    int k = (int)(i / N), n = (int)(i % N);
    WT[(size_t)n * K + k] = (_Float16)W[i];
  }
}

// ---------------------------------------------------------------------------
// C[M,N] = epilogue(A[M,K] @ W + bias), W given transposed as BT[N,K] f16.
// KIND: 0 = bias only, 1 = bias + residual R, 2 = bias + exact GELU.
// Block = 256 thr (8 waves), tile 128x128, K-step 32, wave tile 32x64.
// B tile staged via async global->LDS (ASYNCcnt); A tile converted f32->f16
// on the VALU path; next A tile prefetched to overlap HBM latency.
// ---------------------------------------------------------------------------
template <int KIND>
__global__ __launch_bounds__(256) void gemm_kernel(
    const float* __restrict__ A, const _Float16* __restrict__ BT,
    const float* __restrict__ bias, const float* __restrict__ R,
    float* __restrict__ C, int M, int N, int K) {
  __shared__ __align__(32) _Float16 As[128 * 32];
  __shared__ __align__(32) _Float16 Bs[128 * 32];
  const int tid = threadIdx.x;
  const int lane = tid & 31, wid = tid >> 5;
  const int wm = wid & 3, wn = wid >> 2;
  const int lo = lane & 15, hi = lane >> 4;
  const int m0 = blockIdx.y * 128;
  const int n0 = blockIdx.x * 128;

  v8f acc[2][4];
#pragma unroll
  for (int i = 0; i < 2; i++)
#pragma unroll
    for (int j = 0; j < 4; j++)
#pragma unroll
      for (int r = 0; r < 8; r++) acc[i][j][r] = 0.f;

  for (int k0 = 0; k0 < K; k0 += 32) {
    // Async-stage B tile (128 rows of BT, 32 f16 each): pure 16B copies
#pragma unroll
    for (int i = 0; i < 2; i++) {
      int c = tid + i * 256;
      int n = c >> 2, kq = (c & 3) * 8;
      unsigned lds_off = (unsigned)(uintptr_t)(Bs + n * 32 + kq);
      async_copy_b128(lds_off, BT + (size_t)(n0 + n) * K + k0 + kq);
    }
    // Stage A tile (128x32 f32 -> f16), bounds-checked on M
#pragma unroll
    for (int i = 0; i < 4; i++) {
      int c = tid + i * 256;
      int r = c >> 3, kq = (c & 7) * 4;
      int mg = m0 + r;
      float4 f = make_float4(0.f, 0.f, 0.f, 0.f);
      if (mg < M) {
        f = *(const float4*)(A + (size_t)mg * K + k0 + kq);
        if (k0 + 32 < K)  // keep next A tile in flight (global_prefetch_b8)
          __builtin_prefetch(A + (size_t)mg * K + k0 + 32 + kq, 0, 3);
      }
      v4h hq = {(_Float16)f.x, (_Float16)f.y, (_Float16)f.z, (_Float16)f.w};
      *(v4h*)(As + r * 32 + kq) = hq;
    }
    wait_async0();
    __syncthreads();

    v16h af[2], bf[4];
#pragma unroll
    for (int tm = 0; tm < 2; tm++) {
      const _Float16* p = As + (wm * 32 + tm * 16 + lo) * 32;
      af[tm] = *(const v16h*)(p + 16 * hi);
    }
#pragma unroll
    for (int tn = 0; tn < 4; tn++) {
      const _Float16* p = Bs + (wn * 64 + tn * 16 + lo) * 32 + 16 * hi;
      bf[tn] = *(const v16h*)p;
    }
#pragma unroll
    for (int tm = 0; tm < 2; tm++)
#pragma unroll
      for (int tn = 0; tn < 4; tn++)
        acc[tm][tn] = wmma_f16(af[tm], bf[tn], acc[tm][tn]);
    __syncthreads();
  }

  // Epilogue: C layout elem r -> row r + 8*hi, col lo
#pragma unroll
  for (int tm = 0; tm < 2; tm++) {
#pragma unroll
    for (int tn = 0; tn < 4; tn++) {
      int n = n0 + wn * 64 + tn * 16 + lo;
      float bv = bias[n];
#pragma unroll
      for (int r = 0; r < 8; r++) {
        int m = m0 + wm * 32 + tm * 16 + r + 8 * hi;
        if (m < M) {
          float val = acc[tm][tn][r] + bv;
          if (KIND == 1) val += R[(size_t)m * N + n];
          if (KIND == 2) val = 0.5f * val * (1.0f + erff(val * 0.70710678118f));
          C[(size_t)m * N + n] = val;
        }
      }
    }
  }
}

// ---------------------------------------------------------------------------
// Fused attention, one (b,h) per 2 blocks; 8 waves/block, each wave owns a
// 16-query tile. K,V staged in LDS (f16), online softmax, WMMA for S and P·V.
// q/k/v layout: [B, L, H, HD] f32. O written to [B, L, H, HD] f32.
// ---------------------------------------------------------------------------
__global__ __launch_bounds__(256) void attn_kernel(const float* __restrict__ Qg,
                                                   const float* __restrict__ Kg,
                                                   const float* __restrict__ Vg,
                                                   float* __restrict__ Og) {
  __shared__ __align__(32) _Float16 Klds[Lsz * HDsz];   // [key][hd]
  __shared__ __align__(32) _Float16 Vt[HDsz * KPAD];    // [hd][key], padded
  __shared__ __align__(32) _Float16 Plds[8 * 512];      // per-wave 16x32 P tile
  const int tid = threadIdx.x;
  const int lane = tid & 31, wid = tid >> 5;
  const int lo = lane & 15, hi = lane >> 4;
  const int bh = blockIdx.x >> 1;
  const int b = bh / Hsz, h = bh % Hsz;
  const int qtile = (blockIdx.x & 1) * 8 + wid;   // 0..15, 13..15 idle
  const bool qt_ok = (qtile < 13);
  const int q0 = (qt_ok ? qtile : 12) * 16;

  // Zero V^T pad region (keys 197..223) so WMMA never sees garbage bits
  for (int i = tid; i < HDsz * 32; i += 256) {
    int hd = i >> 5, key = 192 + (i & 31);
    if (key >= Lsz) Vt[hd * KPAD + key] = (_Float16)0.f;
  }
  // Stage K [key][hd] and V^T [hd][key]
  for (int i = tid; i < Lsz * HDsz; i += 256) {
    int key = i >> 6, hd = i & 63;
    size_t gidx = (((size_t)b * Lsz + key) * Hsz + h) * HDsz + hd;
    Klds[i] = (_Float16)Kg[gidx];
    Vt[hd * KPAD + key] = (_Float16)Vg[gidx];
  }
  __syncthreads();

  // Q A-fragments for hd 0..31 and 32..63, loaded once from global
  v16h qf[2];
  {
    int q = q0 + lo;
    bool ok = q < Lsz;
    const float* qp = Qg + (((size_t)b * Lsz + (ok ? q : 0)) * Hsz + h) * HDsz;
#pragma unroll
    for (int kk = 0; kk < 2; kk++) {
      int o1 = kk * 32 + 16 * hi;
      int o2 = o1 + 8;
      float4 f0 = make_float4(0.f, 0.f, 0.f, 0.f), f1 = f0, f2 = f0, f3 = f0;
      if (ok) {
        f0 = *(const float4*)(qp + o1);
        f1 = *(const float4*)(qp + o1 + 4);
        f2 = *(const float4*)(qp + o2);
        f3 = *(const float4*)(qp + o2 + 4);
      }
      v16h t;
      t[0] = (_Float16)f0.x; t[1] = (_Float16)f0.y;
      t[2] = (_Float16)f0.z; t[3] = (_Float16)f0.w;
      t[4] = (_Float16)f1.x; t[5] = (_Float16)f1.y;
      t[6] = (_Float16)f1.z; t[7] = (_Float16)f1.w;
      t[8] = (_Float16)f2.x; t[9] = (_Float16)f2.y;
      t[10] = (_Float16)f2.z; t[11] = (_Float16)f2.w;
      t[12] = (_Float16)f3.x; t[13] = (_Float16)f3.y;
      t[14] = (_Float16)f3.z; t[15] = (_Float16)f3.w;
      qf[kk] = t;
    }
  }

  float mrow[8], lrow[8], corr[8];
  v8f oacc[4];
#pragma unroll
  for (int r = 0; r < 8; r++) { mrow[r] = -1e30f; lrow[r] = 0.f; }
#pragma unroll
  for (int t = 0; t < 4; t++)
#pragma unroll
    for (int r = 0; r < 8; r++) oacc[t][r] = 0.f;
  _Float16* Pb = Plds + wid * 512;

  for (int kt = 0; kt < 7; kt++) {   // 7 chunks of 32 keys cover 197
    const int kt0 = kt * 32;
    v8f s[2];
#pragma unroll
    for (int t = 0; t < 2; t++) {
      int key = kt0 + t * 16 + lo;
      if (key > Lsz - 1) key = Lsz - 1;  // clamp; masked below
      v8f z;
#pragma unroll
      for (int r = 0; r < 8; r++) z[r] = 0.f;
      v16h kf0 = *(const v16h*)(Klds + key * 64 + 16 * hi);
      v16h kf1 = *(const v16h*)(Klds + key * 64 + 32 + 16 * hi);
      z = wmma_f16(qf[0], kf0, z);
      z = wmma_f16(qf[1], kf1, z);
      s[t] = z;
    }
    // online softmax over 32 keys: row r lives on the 16-lane half, col = lo
#pragma unroll
    for (int r = 0; r < 8; r++) {
      float s0 = s[0][r] * 0.125f, s1 = s[1][r] * 0.125f;
      if (kt0 + lo >= Lsz) s0 = -1e30f;
      if (kt0 + 16 + lo >= Lsz) s1 = -1e30f;
      float mx = fmaxf(s0, s1);
      mx = fmaxf(mx, __shfl_xor(mx, 1, 32));
      mx = fmaxf(mx, __shfl_xor(mx, 2, 32));
      mx = fmaxf(mx, __shfl_xor(mx, 4, 32));
      mx = fmaxf(mx, __shfl_xor(mx, 8, 32));
      float nm = fmaxf(mrow[r], mx);
      float p0 = __expf(s0 - nm), p1 = __expf(s1 - nm);
      float ps = p0 + p1;
      ps += __shfl_xor(ps, 1, 32);
      ps += __shfl_xor(ps, 2, 32);
      ps += __shfl_xor(ps, 4, 32);
      ps += __shfl_xor(ps, 8, 32);
      float cr = __expf(mrow[r] - nm);
      lrow[r] = lrow[r] * cr + ps;
      mrow[r] = nm;
      corr[r] = cr;
      int prow = r + 8 * hi;
      Pb[prow * 32 + lo] = (_Float16)p0;
      Pb[prow * 32 + 16 + lo] = (_Float16)p1;
    }
#pragma unroll
    for (int t = 0; t < 4; t++)
#pragma unroll
      for (int r = 0; r < 8; r++) oacc[t][r] *= corr[r];
    // cross-lane LDS RAW within the wave: drain DS counter
    asm volatile("s_wait_dscnt 0" ::: "memory");
    const _Float16* pp = Pb + lo * 32;
    v16h pf = *(const v16h*)(pp + 16 * hi);
#pragma unroll
    for (int t = 0; t < 4; t++) {
      v16h vf = *(const v16h*)(Vt + (t * 16 + lo) * KPAD + kt0 + 16 * hi);
      oacc[t] = wmma_f16(pf, vf, oacc[t]);
    }
  }

#pragma unroll
  for (int r = 0; r < 8; r++) {
    int q = q0 + r + 8 * hi;
    if (qt_ok && q < Lsz) {
      float inv = 1.f / lrow[r];
#pragma unroll
      for (int t = 0; t < 4; t++) {
        Og[(((size_t)b * Lsz + q) * Hsz + h) * HDsz + t * 16 + lo] =
            oacc[t][r] * inv;
      }
    }
  }
}

// ---------------------------------------------------------------------------
// Host orchestration
// ---------------------------------------------------------------------------
extern "C" void kernel_launch(void* const* d_in, const int* in_sizes, int n_in,
                              void* d_out, int out_size, void* d_ws,
                              size_t ws_size, hipStream_t stream) {
  (void)in_sizes; (void)n_in; (void)out_size; (void)ws_size;
  const float* x     = (const float*)d_in[0];
  const float* pos   = (const float*)d_in[1];
  const float* ln1_g = (const float*)d_in[2];
  const float* ln1_b = (const float*)d_in[3];
  const float* Wq    = (const float*)d_in[4];
  const float* bq    = (const float*)d_in[5];
  const float* Wk    = (const float*)d_in[6];
  const float* bk    = (const float*)d_in[7];
  const float* Wv    = (const float*)d_in[8];
  const float* bv    = (const float*)d_in[9];
  const float* Wo    = (const float*)d_in[10];
  const float* bo    = (const float*)d_in[11];
  const float* ln2_g = (const float*)d_in[12];
  const float* ln2_b = (const float*)d_in[13];
  const float* W1    = (const float*)d_in[14];
  const float* b1    = (const float*)d_in[15];
  const float* W2    = (const float*)d_in[16];
  const float* b2    = (const float*)d_in[17];
  const float* lnf_g = (const float*)d_in[18];
  const float* lnf_b = (const float*)d_in[19];

  const size_t MD = (size_t)Msz * Dsz;   // 9,682,944
  const size_t MF = (size_t)Msz * Fsz;   // 38,731,776
  float* hbuf = (float*)d_ws;
  float* ybuf = hbuf + MD;   // LN out; reused as attention output buffer
  float* qb   = ybuf + MD;
  float* kb   = qb + MD;
  float* vb   = kb + MD;
  float* mid  = vb + MD;     // MLP hidden [M, F]
  _Float16* wbuf = (_Float16*)(mid + MF);  // transposed f16 weights, <= D*F

  auto wconv = [&](const float* W, int K, int N) {
    size_t tot = (size_t)K * N;
    wconv_kernel<<<dim3((unsigned)((tot + 255) / 256)), 256, 0, stream>>>(W, wbuf,
                                                                          K, N);
  };

  addpos_kernel<<<dim3((unsigned)((MD + 255) / 256)), 256, 0, stream>>>(x, pos,
                                                                        hbuf);

  const dim3 lngrid((Msz + 7) / 8);
  const dim3 gD(Dsz / 128, (Msz + 127) / 128);
  const dim3 gF(Fsz / 128, (Msz + 127) / 128);
  const size_t dd = (size_t)Dsz * Dsz, df = (size_t)Dsz * Fsz;

  for (int l = 0; l < NLsz; l++) {
    // --- attention sub-block ---
    ln_kernel<<<lngrid, 256, 0, stream>>>(hbuf, ln1_g + (size_t)l * Dsz,
                                          ln1_b + (size_t)l * Dsz, ybuf, Msz);
    wconv(Wq + (size_t)l * dd, Dsz, Dsz);
    gemm_kernel<0><<<gD, 256, 0, stream>>>(ybuf, wbuf, bq + (size_t)l * Dsz,
                                           nullptr, qb, Msz, Dsz, Dsz);
    wconv(Wk + (size_t)l * dd, Dsz, Dsz);
    gemm_kernel<0><<<gD, 256, 0, stream>>>(ybuf, wbuf, bk + (size_t)l * Dsz,
                                           nullptr, kb, Msz, Dsz, Dsz);
    wconv(Wv + (size_t)l * dd, Dsz, Dsz);
    gemm_kernel<0><<<gD, 256, 0, stream>>>(ybuf, wbuf, bv + (size_t)l * Dsz,
                                           nullptr, vb, Msz, Dsz, Dsz);
    attn_kernel<<<dim3(Bsz * Hsz * 2), 256, 0, stream>>>(qb, kb, vb, ybuf);
    wconv(Wo + (size_t)l * dd, Dsz, Dsz);
    gemm_kernel<1><<<gD, 256, 0, stream>>>(ybuf, wbuf, bo + (size_t)l * Dsz,
                                           hbuf, hbuf, Msz, Dsz, Dsz);
    // --- MLP sub-block ---
    ln_kernel<<<lngrid, 256, 0, stream>>>(hbuf, ln2_g + (size_t)l * Dsz,
                                          ln2_b + (size_t)l * Dsz, ybuf, Msz);
    wconv(W1 + (size_t)l * df, Dsz, Fsz);
    gemm_kernel<2><<<gF, 256, 0, stream>>>(ybuf, wbuf, b1 + (size_t)l * Fsz,
                                           nullptr, mid, Msz, Fsz, Dsz);
    wconv(W2 + (size_t)l * df, Fsz, Dsz);
    gemm_kernel<1><<<gD, 256, 0, stream>>>(mid, wbuf, b2 + (size_t)l * Dsz,
                                           hbuf, hbuf, Msz, Dsz, Fsz);
  }
  ln_kernel<<<lngrid, 256, 0, stream>>>(hbuf, lnf_g, lnf_b, (float*)d_out, Msz);
}
